// ShapeWeaverDecoder_78030965833881
// MI455X (gfx1250) — matrix operand order repaired and checked by reading, not compiled, not run on hardware
//
#include <hip/hip_runtime.h>

// ---------------- problem constants ----------------
#define NB 16        // persistent workgroups: one per 16-unit block (16*16 = 256 units)
#define TT 2048      // timesteps
#define BSZ 32       // batch

typedef __attribute__((ext_vector_type(16))) __bf16 bfx16;
typedef __attribute__((ext_vector_type(8)))  float  fx8;
typedef __attribute__((ext_vector_type(4)))  int    v4i;

union AFrag { bfx16 v; v4i q[2]; };

// ---------------- d_ws layout (bytes) ----------------
#define WS_W1   0u          // 16 wgp * 9 kt * 4 gate * 512 bf16  = 589824 B
#define WS_W2   589824u     // 16 wgp * 16 kt * 4 gate * 512 bf16 = 1048576 B
#define WS_H1   1638400u    // h1 broadcast: 32*256 bf16 = 16384 B
#define WS_H2   1654784u    // h2 broadcast: 16384 B
#define WS_PART 1671168u    // Wo partials: 16*32*2 f32 = 4096 B
#define WS_CTR  1675264u    // grid-barrier counter (4 B)

// ---------------- LDS layout (bytes), dynamic shared ----------------
#define LDS_BYTES 170688

__device__ __forceinline__ float sigm(float x) { return 1.f / (1.f + __expf(-x)); }
__device__ __forceinline__ float tanh_(float x) {
  x = fminf(fmaxf(x, -15.f), 15.f);
  float e = __expf(2.f * x);
  return (e - 1.f) / (e + 1.f);
}

// device-scope barrier across the 16 persistent workgroups.
__device__ __forceinline__ void gridbar(unsigned* ctr, unsigned idx) {
  __threadfence();
  __syncthreads();
  if (threadIdx.x == 0) {
    __hip_atomic_fetch_add(ctr, 1u, __ATOMIC_RELEASE, __HIP_MEMORY_SCOPE_AGENT);
    unsigned tgt = (idx + 1u) * (unsigned)NB;
    while (__hip_atomic_load(ctr, __ATOMIC_ACQUIRE, __HIP_MEMORY_SCOPE_AGENT) < tgt)
      __builtin_amdgcn_s_sleep(2);
  }
  __syncthreads();
}

// single WMMA step helper: acc += A(kt) * B(kt)
__device__ __forceinline__ fx8 wmma_kt(const char* aB, const char* bB, int kt, fx8 acc) {
  AFrag a, b;
  a.q[0] = *(const v4i*)(aB + kt * 64);
  a.q[1] = *(const v4i*)(aB + kt * 64 + 16);
  const char* bp = bB + kt * 4096;
  b.q[0] = *(const v4i*)(bp);
  b.q[1] = *(const v4i*)(bp + 16);
  return __builtin_amdgcn_wmma_f32_16x16x32_bf16(false, a.v, false, b.v,
                                                 (short)0, acc, false, false);
}

// ---------------------------------------------------------------------------
// Kernel 0: pack weights into bf16 WMMA B-fragment order, zero barrier counter.
// W1 rows: 0..255 = R1 (h1 input), 256..257 = K1 (xy coords), 258..287 = 0.
// W2 rows: 0..255 = K2 (h1 input), 256..511 = R2 (h2 state).
// ---------------------------------------------------------------------------
__global__ void sw_pack(const float* __restrict__ K1, const float* __restrict__ R1,
                        const float* __restrict__ K2, const float* __restrict__ R2,
                        unsigned char* __restrict__ ws) {
  int tid = blockIdx.x * blockDim.x + threadIdx.x;
  if (tid == 0) *(unsigned*)(ws + WS_CTR) = 0u;
  const int NW1 = 16 * 9 * 4 * 32;    // 18432 lane-fragments
  const int NW2 = 16 * 16 * 4 * 32;   // 32768 lane-fragments
  if (tid < NW1) {
    int lane = tid & 31, g = (tid >> 5) & 3, kt = (tid >> 7) % 9, w = tid / 1152;
    int col = g * 256 + w * 16 + (lane & 15);
    int kbase = kt * 32 + ((lane >> 4) << 4);
    __bf16* dst = (__bf16*)(ws + WS_W1) + tid * 16;
#pragma unroll
    for (int e = 0; e < 16; ++e) {
      int r = kbase + e;
      float v = (r < 256) ? R1[r * 1024 + col]
                          : ((r < 258) ? K1[(r - 256) * 1024 + col] : 0.f);
      dst[e] = (__bf16)v;
    }
  } else if (tid < NW1 + NW2) {
    int t = tid - NW1;
    int lane = t & 31, g = (t >> 5) & 3, kt = (t >> 7) & 15, w = t >> 11;
    int col = g * 256 + w * 16 + (lane & 15);
    int kbase = kt * 32 + ((lane >> 4) << 4);
    __bf16* dst = (__bf16*)(ws + WS_W2) + t * 16;
#pragma unroll
    for (int e = 0; e < 16; ++e) {
      int r = kbase + e;
      float v = (r < 256) ? K2[r * 1024 + col] : R2[(r - 256) * 1024 + col];
      dst[e] = (__bf16)v;
    }
  }
}

// ---------------------------------------------------------------------------
// Kernel 1: persistent 2-layer LSTM decoder. 16 blocks x 256 threads (8 waves).
// Wave (m = wv&1, gate = wv>>1) owns one 16x16 z-tile of its WGP's unit block.
// Dual WMMA accumulators halve the RAW chain; the h2 broadcast gather is
// software-pipelined across the grid barrier in registers so its L2 latency
// hides behind the next step's layer-1 GEMM.
// ---------------------------------------------------------------------------
__global__ __launch_bounds__(256, 1)
void sw_decoder(const float* __restrict__ ctx, const float* __restrict__ Wc,
                const float* __restrict__ bc, const float* __restrict__ b1,
                const float* __restrict__ b2, const float* __restrict__ Wo,
                const float* __restrict__ bo,
                unsigned char* __restrict__ ws, float* __restrict__ out) {
  extern __shared__ char smem[];
  __bf16* ldsW1 = (__bf16*)(smem + 0);
  __bf16* ldsW2 = (__bf16*)(smem + 36864);
  __bf16* ldsA1 = (__bf16*)(smem + 102400);  // [32][288]: h1 | x | pad
  __bf16* ldsA2 = (__bf16*)(smem + 120832);  // [32][512]: h1 | h2
  float*  zbuf  = (float*)(smem + 153600);   // [wave][256]
  float*  c1buf = (float*)(smem + 161792);   // [512]
  float*  c2buf = (float*)(smem + 163840);   // [512]
  float*  pbuf  = (float*)(smem + 165888);   // [32][16][2]
  float*  b1s   = (float*)(smem + 169984);
  float*  b2s   = (float*)(smem + 170240);
  float*  wos   = (float*)(smem + 170496);
  float*  bos   = (float*)(smem + 170624);

  const int tid  = threadIdx.x;
  const int wgp  = blockIdx.x;
  const int lane = tid & 31;
  const int wv   = tid >> 5;
  const int m    = wv & 1;
  const int gate = wv >> 1;

  __bf16*  h1g   = (__bf16*)(ws + WS_H1);
  __bf16*  h2g   = (__bf16*)(ws + WS_H2);
  float*   partG = (float*)(ws + WS_PART);
  unsigned* ctr  = (unsigned*)(ws + WS_CTR);

  // ---- prologue: weight slices global -> LDS (resident for all 2048 steps) ----
  {
    const v4i* s1 = (const v4i*)(ws + WS_W1 + (unsigned)wgp * 36864u);
    v4i* d1 = (v4i*)ldsW1;
    for (int i = tid; i < 2304; i += 256) d1[i] = s1[i];
    const v4i* s2 = (const v4i*)(ws + WS_W2 + (unsigned)wgp * 65536u);
    v4i* d2 = (v4i*)ldsW2;
    for (int i = tid; i < 4096; i += 256) d2[i] = s2[i];
  }
  if (tid < 64) {
    int g = tid >> 4, u = tid & 15;
    b1s[tid] = b1[g * 256 + wgp * 16 + u];
    b2s[tid] = b2[g * 256 + wgp * 16 + u];
  }
  if (tid < 32) wos[tid] = Wo[(wgp * 16 + (tid >> 1)) * 2 + (tid & 1)];
  if (tid < 2)  bos[tid] = bo[tid];
  for (int e = tid; e < 512; e += 256) { c1buf[e] = 0.f; c2buf[e] = 0.f; }
  if (tid < 32) {  // A1 row tid: x0 = 0.5 in cols 256..257, zero pad cols 258..287
    unsigned* row = (unsigned*)((char*)ldsA1 + tid * 576);
    row[128] = 0x3F003F00u;  // two bf16(0.5)
#pragma unroll
    for (int j = 129; j < 144; ++j) row[j] = 0u;
  }
  // h2 gather registers, pipelined across iterations; h2_0 = 0
  v4i h2r[4];
#pragma unroll
  for (int s = 0; s < 4; ++s) { h2r[s].x = 0; h2r[s].y = 0; h2r[s].z = 0; h2r[s].w = 0; }

  // initial h1 = relu(ctx @ Wc + bc): this WGP computes its 16 unit columns
  for (int s = 0; s < 2; ++s) {
    int e = tid + s * 256; int b = e >> 4, u = e & 15;
    float a = bc[wgp * 16 + u];
    const float* cr = ctx + b * 512;
    const float* wc = Wc + wgp * 16 + u;
    for (int k = 0; k < 512; ++k) a += cr[k] * wc[k * 256];
    a = fmaxf(a, 0.f);
    h1g[b * 256 + wgp * 16 + u] = (__bf16)a;
  }
  gridbar(ctr, 0u);
  // gather full h1_0 into A1 cols 0..255
  for (int i = tid; i < 1024; i += 256) {
    int b = i >> 5, c = i & 31;
    v4i v = ((const v4i*)h1g)[i];
    ((v4i*)ldsA1)[b * 36 + c] = v;
  }
  __syncthreads();

  // ------------------------------ time loop ------------------------------
  for (int t = 0; t < TT; ++t) {
    // ---- layer-1 GEMM: z1 = [h1, x] @ W1slice + b1, K = 288 (9 k-tiles) ----
    {
      float bv = b1s[gate * 16 + (lane & 15)];
      fx8 acc0, acc1;
#pragma unroll
      for (int r = 0; r < 8; ++r) { acc0[r] = bv; acc1[r] = 0.f; }
      const char* aB = (const char*)ldsA1 + (16 * m + (lane & 15)) * 576 + ((lane >> 4) << 5);
      const char* bB = (const char*)ldsW1 + gate * 1024 + lane * 32;
#pragma unroll
      for (int kt = 0; kt < 8; kt += 2) {     // two independent WMMA chains
        acc0 = wmma_kt(aB, bB, kt,     acc0);
        acc1 = wmma_kt(aB, bB, kt + 1, acc1);
      }
      acc0 = wmma_kt(aB, bB, 8, acc0);        // odd leftover tile
      int hi = lane >> 4, n = lane & 15;
#pragma unroll
      for (int r = 0; r < 8; ++r)
        zbuf[wv * 256 + (r + 8 * hi) * 16 + n] = acc0[r] + acc1[r];
    }
    // drain pipelined h2 gather into A2 cols 256..511 (loads issued before the
    // layer-1 GEMM above; their latency is hidden behind it)
#pragma unroll
    for (int s = 0; s < 4; ++s) {
      int i = tid + s * 256;
      int b = i >> 5, c = i & 31;
      ((v4i*)ldsA2)[b * 64 + 32 + c] = h2r[s];
    }
    __syncthreads();
    // ---- layer-1 cell update, publish h1 slice ----
    for (int s = 0; s < 2; ++s) {
      int e = tid + s * 256; int b = e >> 4, u = e & 15;
      int mm = b >> 4, idx = (b & 15) * 16 + u;
      float zi = zbuf[(0 * 2 + mm) * 256 + idx];
      float zf = zbuf[(1 * 2 + mm) * 256 + idx];
      float zg = zbuf[(2 * 2 + mm) * 256 + idx];
      float zo = zbuf[(3 * 2 + mm) * 256 + idx];
      float c  = c1buf[e];
      float cn = sigm(zf) * c + sigm(zi) * tanh_(zg);
      float h  = sigm(zo) * tanh_(cn);
      c1buf[e] = cn;
      h1g[b * 256 + wgp * 16 + u] = (__bf16)h;
    }
    gridbar(ctr, 1u + 2u * (unsigned)t);
    // gather full h1_t -> A2 cols 0..255 (layer-2 input) and A1 cols 0..255 (next step)
    for (int i = tid; i < 1024; i += 256) {
      int b = i >> 5, c = i & 31;
      v4i v = ((const v4i*)h1g)[i];
      ((v4i*)ldsA2)[b * 64 + c] = v;
      ((v4i*)ldsA1)[b * 36 + c] = v;
    }
    __syncthreads();
    // ---- layer-2 GEMM: z2 = [h1, h2] @ W2slice + b2, K = 512 (16 k-tiles) ----
    {
      float bv = b2s[gate * 16 + (lane & 15)];
      fx8 acc0, acc1;
#pragma unroll
      for (int r = 0; r < 8; ++r) { acc0[r] = bv; acc1[r] = 0.f; }
      const char* aB = (const char*)ldsA2 + (16 * m + (lane & 15)) * 1024 + ((lane >> 4) << 5);
      const char* bB = (const char*)ldsW2 + gate * 1024 + lane * 32;
#pragma unroll
      for (int kt = 0; kt < 16; kt += 2) {    // two independent WMMA chains
        acc0 = wmma_kt(aB, bB, kt,     acc0);
        acc1 = wmma_kt(aB, bB, kt + 1, acc1);
      }
      int hi = lane >> 4, n = lane & 15;
#pragma unroll
      for (int r = 0; r < 8; ++r)
        zbuf[wv * 256 + (r + 8 * hi) * 16 + n] = acc0[r] + acc1[r];
    }
    __syncthreads();
    // ---- layer-2 cell update, publish h2 slice + Wo partial dots ----
    for (int s = 0; s < 2; ++s) {
      int e = tid + s * 256; int b = e >> 4, u = e & 15;
      int mm = b >> 4, idx = (b & 15) * 16 + u;
      float zi = zbuf[(0 * 2 + mm) * 256 + idx];
      float zf = zbuf[(1 * 2 + mm) * 256 + idx];
      float zg = zbuf[(2 * 2 + mm) * 256 + idx];
      float zo = zbuf[(3 * 2 + mm) * 256 + idx];
      float c  = c2buf[e];
      float cn = sigm(zf) * c + sigm(zi) * tanh_(zg);
      float h  = sigm(zo) * tanh_(cn);
      c2buf[e] = cn;
      h2g[b * 256 + wgp * 16 + u] = (__bf16)h;
      pbuf[e * 2 + 0] = h * wos[u * 2 + 0];
      pbuf[e * 2 + 1] = h * wos[u * 2 + 1];
    }
    __syncthreads();
    if (tid < 64) {
      int b = tid >> 1, k = tid & 1;
      float sum = 0.f;
#pragma unroll
      for (int u = 0; u < 16; ++u) sum += pbuf[(b * 16 + u) * 2 + k];
      partG[(wgp * 32 + b) * 2 + k] = sum;
    }
    gridbar(ctr, 2u + 2u * (unsigned)t);
    // issue h2 gather into registers; consumed after next step's layer-1 GEMM
#pragma unroll
    for (int s = 0; s < 4; ++s) {
      int i = tid + s * 256;
      h2r[s] = ((const v4i*)h2g)[i];
    }
    // out_t = sigmoid(h2 @ Wo + bo); it is also x_{t+1}
    if (tid < 64) {
      int b = tid >> 1, k = tid & 1;
      float sum = bos[k];
      for (int w2 = 0; w2 < NB; ++w2) sum += partG[(w2 * 32 + b) * 2 + k];
      float x = sigm(sum);
      ldsA1[b * 288 + 256 + k] = (__bf16)x;
      if (wgp == 0) out[(b * TT + t) * 2 + k] = x;
    }
    __syncthreads();
  }
}

// ---------------------------------------------------------------------------
extern "C" void kernel_launch(void* const* d_in, const int* in_sizes, int n_in,
                              void* d_out, int out_size, void* d_ws, size_t ws_size,
                              hipStream_t stream) {
  const float* ctx = (const float*)d_in[0];
  const float* Wc  = (const float*)d_in[1];
  const float* bc  = (const float*)d_in[2];
  const float* K1  = (const float*)d_in[3];
  const float* R1  = (const float*)d_in[4];
  const float* b1  = (const float*)d_in[5];
  const float* K2  = (const float*)d_in[6];
  const float* R2  = (const float*)d_in[7];
  const float* b2  = (const float*)d_in[8];
  const float* Wo  = (const float*)d_in[9];
  const float* bo  = (const float*)d_in[10];
  unsigned char* ws = (unsigned char*)d_ws;
  float* out = (float*)d_out;

  (void)in_sizes; (void)n_in; (void)out_size; (void)ws_size;

  // allow > 64KB dynamic LDS (gfx1250 WGP has 320KB); host-side, not captured
  hipFuncSetAttribute((const void*)sw_decoder,
                      hipFuncAttributeMaxDynamicSharedMemorySize, LDS_BYTES);

  // Phase 1: pack weights into WMMA fragment order + reset grid-barrier counter
  sw_pack<<<200, 256, 0, stream>>>(K1, R1, K2, R2, ws);
  // Phase 2: persistent decoder, 16 co-resident workgroups, full 2048-step recurrence
  sw_decoder<<<NB, 256, LDS_BYTES, stream>>>(ctx, Wc, bc, b1, b2, Wo, bo, ws, out);
}
